// CapsuleNetwork_79370995630139
// MI455X (gfx1250) — matrix operand order repaired and checked by reading, not compiled, not run on hardware
//
#include <hip/hip_runtime.h>
#include <hip/hip_bf16.h>

typedef __attribute__((ext_vector_type(16))) __bf16 v16bf;
typedef __attribute__((ext_vector_type(8)))  __bf16 v8bf;
typedef __attribute__((ext_vector_type(8)))  float  v8f;
typedef __attribute__((ext_vector_type(4)))  int    v4i;

#define PI_F 3.14159265358979323846f

// ---------------- dims ----------------
constexpr int BB    = 64;
constexpr int NPRIM = 4608;
constexpr int NSEC  = 10;
constexpr int KPRIM = 81 * 256;         // 20736
constexpr int MPRIM = BB * 12 * 12;     // 9216
constexpr int NW_PRIM = 9 * 9 * 256 * 1024;

// ---------------- CDNA5 async global->LDS ----------------
#if defined(__has_builtin)
# if __has_builtin(__builtin_amdgcn_global_load_async_to_lds_b128)
#  define HAVE_ASYNC 1
# endif
#endif
#ifndef HAVE_ASYNC
# define HAVE_ASYNC 0
#endif

#if HAVE_ASYNC
typedef __attribute__((address_space(1))) v4i* g4_ptr;   // global int4*
typedef __attribute__((address_space(3))) v4i* l4_ptr;   // LDS int4*
__device__ __forceinline__ void async_b128(const __bf16* g, __bf16* l) {
  __builtin_amdgcn_global_load_async_to_lds_b128((g4_ptr)(void*)g, (l4_ptr)(void*)l, 0, 0);
}
__device__ __forceinline__ void wait_async0() {
# if __has_builtin(__builtin_amdgcn_s_wait_asynccnt)
  __builtin_amdgcn_s_wait_asynccnt(0);
# else
  asm volatile("s_wait_asynccnt 0" ::: "memory");
# endif
}
#endif

// ---------------- prep: transpose prim_w (20736 x 1024) f32 -> (1024 x 20736) bf16 ----------------
__global__ __launch_bounds__(256) void k_wT(const float* __restrict__ src, __bf16* __restrict__ dst) {
  __shared__ float tile[32][33];
  const int k0 = blockIdx.x * 32;       // 648
  const int n0 = blockIdx.y * 32;       // 32
  const int tr = threadIdx.x >> 5;      // 0..7
  const int tc = threadIdx.x & 31;
  #pragma unroll
  for (int q = 0; q < 4; ++q) {
    const int r = q * 8 + tr;
    tile[r][tc] = src[(size_t)(k0 + r) * 1024 + n0 + tc];
  }
  __syncthreads();
  #pragma unroll
  for (int q = 0; q < 4; ++q) {
    const int r = q * 8 + tr;           // n within tile
    dst[(size_t)(n0 + r) * (size_t)KPRIM + k0 + tc] = (__bf16)tile[tc][r];
  }
}

// ---------------- conv1: 9x9 VALID, 1->256, relu, bf16 out ----------------
__global__ __launch_bounds__(256) void k_conv1(const float* __restrict__ x,
                                               const float* __restrict__ w,
                                               const float* __restrict__ bias,
                                               __bf16* __restrict__ out) {
  __shared__ float patch[81];
  int blk = blockIdx.x;
  const int ow = blk % 24; blk /= 24;
  const int oh = blk % 24; blk /= 24;
  const int b  = blk;
  const int t = threadIdx.x;            // co
  if (t < 81) {
    const int ky = t / 9, kx = t % 9;
    patch[t] = x[(b * 32 + oh + ky) * 32 + ow + kx];
  }
  __syncthreads();
  float acc = bias[t];
  #pragma unroll
  for (int i = 0; i < 81; ++i) acc = fmaf(patch[i], w[i * 256 + t], acc);
  acc = fmaxf(acc, 0.0f);
  out[((b * 24 + oh) * 24 + ow) * 256 + t] = (__bf16)acc;
}

// ---------------- prim conv implicit GEMM, WMMA bf16, double-buffered LDS ----------------
// M = 9216 (b,oh,ow), N = 1024 (co), K = 20736 ((ky,kx),ci)
constexpr int LDA = 40;
constexpr int LDB = 40;
constexpr int NKCH = KPRIM / 32;        // 648

__global__ __launch_bounds__(256) void k_prim_gemm(const __bf16* __restrict__ A,   // x1 (64,24,24,256) bf16
                                                   const __bf16* __restrict__ WT,  // (1024,20736) bf16
                                                   const float* __restrict__ bias,
                                                   float* __restrict__ out) {      // x2 (9216,1024) f32
  __shared__ __bf16 sA[2][128 * LDA];
  __shared__ __bf16 sB[2][128 * LDB];

  const int tid  = threadIdx.x;
  const int lane = tid & 31;
  const int wave = tid >> 5;
  const int wm = wave & 3;              // 4 m-subtiles of 32 rows
  const int wn = wave >> 2;             // 2 n-subtiles of 64 cols
  const int m0 = blockIdx.x * 128;      // 72
  const int n0 = blockIdx.y * 128;      // 8

  // loader geometry: each thread owns one 16-element half of one row (A and B)
  const int r2 = tid >> 1;              // 0..127
  const int h2 = tid & 1;

  const int gm = m0 + r2;
  const int ab = gm / 144, arem = gm % 144;
  const int aoh = arem / 12, aow = arem % 12;
  const __bf16* bsrc0 = WT + (size_t)(n0 + r2) * (size_t)KPRIM + h2 * 16;

  auto a_src = [&](int kc) -> const __bf16* {
    const int k0 = kc * 32;
    const int pos = k0 >> 8, ci0 = k0 & 255;
    const int ky = pos / 9, kx = pos % 9;
    const int ih = aoh * 2 - 3 + ky;    // SAME pad_lo = 3
    const int iw = aow * 2 - 3 + kx;
    if ((unsigned)ih >= 24u || (unsigned)iw >= 24u) return (const __bf16*)0;
    return A + (((size_t)(ab * 24 + ih) * 24 + iw) * 256 + ci0 + h2 * 16);
  };

  v8f acc[2][4] = {};
  v8bf zz;
  #pragma unroll
  for (int e = 0; e < 8; ++e) zz[e] = (__bf16)0.0f;

#if HAVE_ASYNC
  auto issue = [&](int kc, int buf) {
    __bf16* da = &sA[buf][r2 * LDA + h2 * 16];
    __bf16* db = &sB[buf][r2 * LDB + h2 * 16];
    const __bf16* sa = a_src(kc);
    if (sa) {
      async_b128(sa, da);
      async_b128(sa + 8, da + 8);
    } else {
      ((v8bf*)da)[0] = zz;
      ((v8bf*)da)[1] = zz;
    }
    const __bf16* sb = bsrc0 + kc * 32;
    async_b128(sb, db);
    async_b128(sb + 8, db + 8);
  };
  issue(0, 0);
#else
  v8bf ra0, ra1, rb0, rb1;
  auto gload = [&](int kc) {
    const __bf16* sa = a_src(kc);
    if (sa) { ra0 = ((const v8bf*)sa)[0]; ra1 = ((const v8bf*)sa)[1]; }
    else    { ra0 = zz; ra1 = zz; }
    const __bf16* sb = bsrc0 + kc * 32;
    rb0 = ((const v8bf*)sb)[0];
    rb1 = ((const v8bf*)sb)[1];
  };
  gload(0);
#endif

  for (int kc = 0; kc < NKCH; ++kc) {
    const int buf = kc & 1;
#if HAVE_ASYNC
    wait_async0();                       // my chunk-kc portion landed in LDS
    __syncthreads();                     // whole tile visible; prior readers of buf^1 done
    if (kc + 1 < NKCH) issue(kc + 1, buf ^ 1);
#else
    {
      v8bf* da = (v8bf*)&sA[buf][r2 * LDA + h2 * 16];
      v8bf* db = (v8bf*)&sB[buf][r2 * LDB + h2 * 16];
      da[0] = ra0; da[1] = ra1;
      db[0] = rb0; db[1] = rb1;
    }
    __syncthreads();
    if (kc + 1 < NKCH) {
      const __bf16* sa = a_src(kc + 1);
      if (sa) __builtin_prefetch(sa, 0, 0);
      __builtin_prefetch(bsrc0 + (kc + 1) * 32, 0, 0);
      gload(kc + 1);
    }
#endif

    // fragments per ISA 16-bit A 16x32 / B 32x16 lane striping
    v16bf afr[2], bfr[4];
    {
      const int am = lane & 15;
      const int kh = (lane >> 4) << 3;   // 0 or 8
      #pragma unroll
      for (int i = 0; i < 2; ++i) {
        const __bf16* p = &sA[buf][(wm * 32 + i * 16 + am) * LDA];
        v8bf lo = *(const v8bf*)(p + kh);
        v8bf hi = *(const v8bf*)(p + 16 + kh);
        #pragma unroll
        for (int e = 0; e < 8; ++e) { afr[i][e] = lo[e]; afr[i][8 + e] = hi[e]; }
      }
      const int bn = lane & 15;
      const int kb = (lane >> 4) << 3;   // 0 or 8
      #pragma unroll
      for (int j = 0; j < 4; ++j) {
        const __bf16* p = &sB[buf][(wn * 64 + j * 16 + bn) * LDB + kb];
        v8bf lo = *(const v8bf*)p;
        v8bf hi = *(const v8bf*)(p + 16);
        #pragma unroll
        for (int e = 0; e < 8; ++e) { bfr[j][e] = lo[e]; bfr[j][8 + e] = hi[e]; }
      }
    }

    #pragma unroll
    for (int i = 0; i < 2; ++i)
      #pragma unroll
      for (int j = 0; j < 4; ++j)
        acc[i][j] = __builtin_amdgcn_wmma_f32_16x16x32_bf16(
            false, afr[i], false, bfr[j], (short)0, acc[i][j], false, false);
  }

  // epilogue: C/D layout -> row = (lane>>4)*8 + r, col = lane&15
  const int crow = (lane >> 4) << 3;
  const int ccol = lane & 15;
  #pragma unroll
  for (int i = 0; i < 2; ++i)
    #pragma unroll
    for (int j = 0; j < 4; ++j) {
      const int gn = n0 + wn * 64 + j * 16 + ccol;
      const float bv = bias[gn];
      #pragma unroll
      for (int r = 0; r < 8; ++r) {
        const int gmr = m0 + wm * 32 + i * 16 + crow + r;
        out[(size_t)gmr * 1024 + gn] = acc[i][j][r] + bv;
      }
    }
}

// ---------------- gabor 5x5 depthwise, SAME, bf16 out (== u) ----------------
__global__ __launch_bounds__(256) void k_gabor(const float* __restrict__ x2, __bf16* __restrict__ u) {
  __shared__ float g[25];
  const int t = threadIdx.x;
  if (t < 25) {
    const float freq = 0.7f, theta = 0.9f;
    const float sigma = (1.0f / PI_F) * sqrtf(logf(2.0f) * 0.5f) * 3.0f / freq;
    const int dy = t / 5 - 2, dx = t % 5 - 2;
    const float ct = cosf(theta), st = sinf(theta);
    const float rotx =  dx * ct + dy * st;
    const float roty = -dx * st + dy * ct;
    const float gg = expf(-0.5f * (rotx * rotx + roty * roty) / (sigma * sigma)) /
                     (2.0f * PI_F * sigma * sigma);
    g[t] = gg * cosf(2.0f * PI_F * freq * rotx);
  }
  __syncthreads();
  int blk = blockIdx.x;
  const int ow = blk % 12; blk /= 12;
  const int oh = blk % 12; blk /= 12;
  const int b  = blk;
  for (int q = 0; q < 4; ++q) {
    const int c = t + q * 256;
    float acc = 0.0f;
    #pragma unroll
    for (int ky = 0; ky < 5; ++ky) {
      const int ih = oh + ky - 2;
      if ((unsigned)ih >= 12u) continue;
      #pragma unroll
      for (int kx = 0; kx < 5; ++kx) {
        const int iw = ow + kx - 2;
        if ((unsigned)iw >= 12u) continue;
        acc = fmaf(x2[((size_t)(b * 12 + ih) * 12 + iw) * 1024 + c], g[ky * 5 + kx], acc);
      }
    }
    u[((size_t)(b * 12 + oh) * 12 + ow) * 1024 + c] = (__bf16)acc;
  }
}

// ---------------- u_hat: per-n GEMM 64x160x32 via WMMA ----------------
__global__ __launch_bounds__(256) void k_uhat(const __bf16* __restrict__ u,     // (64,4608,32) bf16
                                              const float* __restrict__ wpose,  // (4608,10,16,32) f32
                                              float* __restrict__ uhat) {       // (64,4608,10,16) f32
  const int lane = threadIdx.x & 31;
  const int wave = threadIdx.x >> 5;
  const int n = blockIdx.x * 8 + wave;

  v16bf afr[4];
  const int am = lane & 15;
  const int kh = (lane >> 4) << 3;
  #pragma unroll
  for (int i = 0; i < 4; ++i) {
    const __bf16* p = u + ((size_t)(i * 16 + am) * 4608 + n) * 32;
    v8bf lo = *(const v8bf*)(p + kh);
    v8bf hi = *(const v8bf*)(p + 16 + kh);
    #pragma unroll
    for (int e = 0; e < 8; ++e) { afr[i][e] = lo[e]; afr[i][8 + e] = hi[e]; }
  }
  const int bcol = lane & 15;           // d
  const int kb = (lane >> 4) << 3;      // p 0 / 8
  const int crow = (lane >> 4) << 3;
  for (int s = 0; s < NSEC; ++s) {
    const float* wp = wpose + (((size_t)n * 10 + s) * 16 + bcol) * 32 + kb;
    v16bf bfr;
    #pragma unroll
    for (int e = 0; e < 16; ++e) bfr[e] = (__bf16)wp[(e & 7) + ((e >> 3) << 4)];
    v8f acc[4] = {};
    #pragma unroll
    for (int i = 0; i < 4; ++i)
      acc[i] = __builtin_amdgcn_wmma_f32_16x16x32_bf16(
          false, afr[i], false, bfr, (short)0, acc[i], false, false);
    #pragma unroll
    for (int i = 0; i < 4; ++i)
      #pragma unroll
      for (int r = 0; r < 8; ++r) {
        const int b = i * 16 + crow + r;
        uhat[(((size_t)b * 4608 + n) * 10 + s) * 16 + bcol] = acc[i][r];
      }
  }
}

// ---------------- routing ----------------
__global__ void k_zero(float* __restrict__ p, int n) {
  const int i = blockIdx.x * blockDim.x + threadIdx.x;
  if (i < n) p[i] = 0.0f;
}

__global__ __launch_bounds__(160) void k_route(const float* __restrict__ uhat,
                                               const float* __restrict__ vprev,
                                               float* __restrict__ logits,
                                               float* __restrict__ svec,
                                               int iter) {
  __shared__ float vsh[160];
  __shared__ float prod[160];
  __shared__ float lg[10];
  const int t = threadIdx.x;            // t = s*16 + d
  const int s = t >> 4;
  const int b = blockIdx.y;
  const int nc = blockIdx.x;            // 18 chunks of 256 n
  if (iter > 0) vsh[t] = vprev[b * 160 + t];
  __syncthreads();
  float acc = 0.0f;
  for (int nn = 0; nn < 256; ++nn) {
    const int n = nc * 256 + nn;
    const float uh = uhat[((size_t)b * 4608 + n) * 160 + t];
    float c;
    if (iter == 0) {
      c = 0.1f;
    } else {
      prod[t] = uh * vsh[t];
      __syncthreads();
      if (t < 10) {
        float a = 0.0f;
        #pragma unroll
        for (int dd = 0; dd < 16; ++dd) a += prod[t * 16 + dd];
        float nl = (iter == 1 ? 0.0f : logits[((size_t)b * 4608 + n) * 10 + t]) + a;
        if (iter == 1) logits[((size_t)b * 4608 + n) * 10 + t] = nl;
        lg[t] = nl;
      }
      __syncthreads();
      float mx = lg[0];
      #pragma unroll
      for (int ss = 1; ss < 10; ++ss) mx = fmaxf(mx, lg[ss]);
      float sum = 0.0f;
      #pragma unroll
      for (int ss = 0; ss < 10; ++ss) sum += __expf(lg[ss] - mx);
      c = __expf(lg[s] - mx) / sum;
      __syncthreads();
    }
    acc = fmaf(c, uh, acc);
  }
  atomicAdd(&svec[b * 160 + t], acc);
}

__global__ __launch_bounds__(160) void k_squash(const float* __restrict__ svec,
                                                float* __restrict__ vout,
                                                float* __restrict__ dout_v,
                                                int write_out) {
  __shared__ float sq[160];
  __shared__ float nrm[10];
  const int t = threadIdx.x, s = t >> 4;
  const int b = blockIdx.x;
  const float sv = svec[b * 160 + t];
  sq[t] = sv * sv;
  __syncthreads();
  if (t < 10) {
    float a = 0.0f;
    #pragma unroll
    for (int dd = 0; dd < 16; ++dd) a += sq[t * 16 + dd];
    nrm[t] = sqrtf(a);
  }
  __syncthreads();
  const float nv = nrm[s];
  const float sqn = nv * nv;
  const float v = (sqn / (1.0f + sqn)) * (sv / (nv + 1e-7f));
  vout[b * 160 + t] = v;
  if (write_out) dout_v[b * 160 + t] = v;
}

// ---------------- decoder ----------------
__global__ void k_mask(const float* __restrict__ v, const float* __restrict__ y,
                       float* __restrict__ vm) {
  const int i = blockIdx.x * blockDim.x + threadIdx.x;
  if (i < 64 * 160) {
    const int b = i / 160, s = (i % 160) >> 4;
    vm[i] = v[i] * y[b * 10 + s];
  }
}

__global__ __launch_bounds__(256) void k_dense(const float* __restrict__ in,
                                               const float* __restrict__ W,
                                               const float* __restrict__ bias,
                                               float* __restrict__ out,
                                               int K, int N, int act) {
  __shared__ float insh[512];
  const int b = blockIdx.y;
  const int j = blockIdx.x * blockDim.x + threadIdx.x;
  for (int k = threadIdx.x; k < K; k += blockDim.x) insh[k] = in[b * K + k];
  __syncthreads();
  if (j < N) {
    float acc = bias[j];
    for (int k = 0; k < K; ++k) acc = fmaf(insh[k], W[(size_t)k * N + j], acc);
    if (act == 0) acc = fmaxf(acc, 0.0f);
    else          acc = 1.0f / (1.0f + __expf(-acc));
    out[(size_t)b * N + j] = acc;
  }
}

// ---------------- launch ----------------
extern "C" void kernel_launch(void* const* d_in, const int* in_sizes, int n_in,
                              void* d_out, int out_size, void* d_ws, size_t ws_size,
                              hipStream_t stream) {
  const float* input_x = (const float*)d_in[0];
  const float* y       = (const float*)d_in[1];
  const float* conv1_w = (const float*)d_in[2];
  const float* conv1_b = (const float*)d_in[3];
  const float* prim_w  = (const float*)d_in[4];
  const float* prim_b  = (const float*)d_in[5];
  const float* w_pose  = (const float*)d_in[6];
  const float* d1_w    = (const float*)d_in[7];
  const float* d1_b    = (const float*)d_in[8];
  const float* d2_w    = (const float*)d_in[9];
  const float* d2_b    = (const float*)d_in[10];
  const float* d3_w    = (const float*)d_in[11];
  const float* d3_b    = (const float*)d_in[12];
  float* out_v = (float*)d_out;            // (64,1,10,16)
  float* out_r = (float*)d_out + 10240;    // (64,1024)

  char* ws = (char*)d_ws;
  auto alloc = [&](size_t bytes) -> void* {
    void* p = (void*)ws;
    ws += (bytes + 255) & ~(size_t)255;
    return p;
  };
  __bf16* wbT    = (__bf16*)alloc((size_t)NW_PRIM * 2);            // 42.5 MB (1024 x 20736)
  __bf16* x1     = (__bf16*)alloc((size_t)64 * 24 * 24 * 256 * 2); // 18.9 MB
  float*  x2     = (float*) alloc((size_t)MPRIM * 1024 * 4);       // 37.7 MB
  __bf16* u      = (__bf16*)alloc((size_t)64 * NPRIM * 32 * 2);    // 18.9 MB
  float*  uhat   = (float*) alloc((size_t)64 * NPRIM * 160 * 4);   // 188.7 MB
  float*  logits = (float*) alloc((size_t)64 * NPRIM * 10 * 4);    // 11.8 MB
  float*  svec   = (float*) alloc(10240 * 4);
  float*  vbuf   = (float*) alloc(10240 * 4);
  float*  vm     = (float*) alloc(10240 * 4);
  float*  r1     = (float*) alloc((size_t)64 * 256 * 4);
  float*  r2     = (float*) alloc((size_t)64 * 512 * 4);

  k_wT<<<dim3(648, 32), 256, 0, stream>>>(prim_w, wbT);
  k_conv1<<<64 * 24 * 24, 256, 0, stream>>>(input_x, conv1_w, conv1_b, x1);
  k_prim_gemm<<<dim3(72, 8), 256, 0, stream>>>(x1, wbT, prim_b, x2);
  k_gabor<<<64 * 12 * 12, 256, 0, stream>>>(x2, u);
  k_uhat<<<576, 256, 0, stream>>>(u, w_pose, uhat);

  for (int iter = 0; iter < 3; ++iter) {
    k_zero<<<40, 256, 0, stream>>>(svec, 10240);
    k_route<<<dim3(18, 64), 160, 0, stream>>>(uhat, vbuf, logits, svec, iter);
    k_squash<<<64, 160, 0, stream>>>(svec, vbuf, out_v, iter == 2 ? 1 : 0);
  }

  k_mask<<<40, 256, 0, stream>>>(vbuf, y, vm);
  k_dense<<<dim3(1, 64), 256, 0, stream>>>(vm, d1_w, d1_b, r1, 160, 256, 0);
  k_dense<<<dim3(2, 64), 256, 0, stream>>>(r1, d2_w, d2_b, r2, 256, 512, 0);
  k_dense<<<dim3(4, 64), 256, 0, stream>>>(r2, d3_w, d3_b, out_r, 512, 1024, 1);
}
